// LSTM_66013647339934
// MI455X (gfx1250) — matrix operand order repaired and checked by reading, not compiled, not run on hardware
//
#include <hip/hip_runtime.h>
#include <hip/hip_bf16.h>
#include <cstdint>
#include <cstddef>

// Problem sizes (fixed by the reference)
#define B_    128
#define T_    256
#define V_    128
#define H_    1024
#define KDIM  1152              // V_ + H_  (fused [x_t ; h] K dimension)
#define G4H   4096              // 4 * H_
#define NWG   128               // persistent workgroups (8 batch-blocks x 16 H-slices)
#define NTHR  256               // 8 waves of 32 (2 waves per gate)

typedef __bf16 bf16;
typedef __bf16 v16bf __attribute__((ext_vector_type(16)));
typedef float  v8f   __attribute__((ext_vector_type(8)));
typedef unsigned int u32x4 __attribute__((ext_vector_type(4)));
typedef int          i32x4 __attribute__((ext_vector_type(4)));
typedef int          i32x8 __attribute__((ext_vector_type(8)));

#if defined(__has_builtin)
#  if __has_builtin(__builtin_amdgcn_tensor_load_to_lds)
#    define HAVE_TDM 1
#  endif
#  if __has_builtin(__builtin_amdgcn_sched_group_barrier)
#    define HAVE_SGB 1
#  endif
#endif

union ABf { v16bf v; uint4 q[2]; };

static __device__ __forceinline__ v8f wmma_bf16(v16bf a, v16bf b, v8f c) {
  // D = A(16x32 bf16) * B(32x16 bf16) + C(16x16 f32)
  return __builtin_amdgcn_wmma_f32_16x16x32_bf16(false, a, false, b, (short)0, c,
                                                 false, false);
}

static __device__ __forceinline__ void kstep_sched_hint() {
#if defined(HAVE_SGB)
  // per k-step: 4 VMEM reads (B frags), 2 DS reads (A frag), 2 matrix ops
  __builtin_amdgcn_sched_group_barrier(0x020, 4, 0);  // VMEM read
  __builtin_amdgcn_sched_group_barrier(0x100, 2, 0);  // DS read
  __builtin_amdgcn_sched_group_barrier(0x008, 2, 0);  // MFMA/WMMA
#endif
}

// ---------------------------------------------------------------------------
// Prep 1: fused bf16 weight, column-major: Wf[col][k], col = g*H + j,
// k: 0..127 = Wx[g][v][j], 128..1151 = Wh[g][r][j].
// ---------------------------------------------------------------------------
__global__ void prep_weights(const float* __restrict__ Wx,
                             const float* __restrict__ Wh,
                             bf16* __restrict__ Wf) {
  size_t idx = (size_t)blockIdx.x * blockDim.x + threadIdx.x;
  if (idx >= (size_t)G4H * KDIM) return;
  int col = (int)(idx / KDIM);
  int kk  = (int)(idx % KDIM);
  int g   = col >> 10;
  int j   = col & 1023;
  float v;
  if (kk < V_) v = Wx[((size_t)g * V_ + kk) * H_ + j];
  else         v = Wh[((size_t)g * H_ + (kk - V_)) * H_ + j];
  Wf[idx] = (bf16)v;
}

// ---------------------------------------------------------------------------
// Prep 2: x -> bf16 (feeds the per-step TDM x-tile loads)
// ---------------------------------------------------------------------------
__global__ void prep_x(const float* __restrict__ x, bf16* __restrict__ xbf) {
  size_t idx = (size_t)blockIdx.x * blockDim.x + threadIdx.x;
  if (idx < (size_t)B_ * T_ * V_) xbf[idx] = (bf16)x[idx];
}

// ---------------------------------------------------------------------------
// Prep 3: bias = bx + bh ; Wl -> bf16 ([v][h] is already column-major B);
// h0 -> bf16 ping buffer; c working copy; reset grid-barrier counter.
// ---------------------------------------------------------------------------
__global__ void prep_misc(const float* __restrict__ bx, const float* __restrict__ bh,
                          const float* __restrict__ Wl, const float* __restrict__ h0,
                          const float* __restrict__ c0,
                          float* __restrict__ bias, bf16* __restrict__ Wlt,
                          bf16* __restrict__ hb0, float* __restrict__ cb,
                          int* __restrict__ cnt) {
  int idx = blockIdx.x * blockDim.x + threadIdx.x;   // 0 .. 131071
  if (idx < G4H)  bias[idx] = bx[idx] + bh[idx];
  if (idx < V_ * H_) Wlt[idx] = (bf16)Wl[idx];
  if (idx < B_ * H_) { hb0[idx] = (bf16)h0[idx]; cb[idx] = c0[idx]; }
  if (idx == 0) *cnt = 0;
}

// ---------------------------------------------------------------------------
// Device-wide barrier: release fence -> arrive -> spin (s_sleep) -> acquire.
// ---------------------------------------------------------------------------
static __device__ __forceinline__ void grid_barrier(int* cnt, int target) {
  __builtin_amdgcn_fence(__ATOMIC_RELEASE, "agent");
  __syncthreads();
  if (threadIdx.x == 0) {
    __hip_atomic_fetch_add(cnt, 1, __ATOMIC_RELAXED, __HIP_MEMORY_SCOPE_AGENT);
    while (__hip_atomic_load(cnt, __ATOMIC_RELAXED, __HIP_MEMORY_SCOPE_AGENT) < target)
      __builtin_amdgcn_s_sleep(2);
  }
  __syncthreads();
  __builtin_amdgcn_fence(__ATOMIC_ACQUIRE, "agent");
}

#if defined(HAVE_TDM)
// Issue a TDM load: g0 built by caller, g1 describes tile shape.
static __device__ __forceinline__ void tdm_load(u32x4 g0, i32x8 g1) {
  i32x4 z4 = {};
#if __clang_major__ >= 23
  i32x8 z8 = {};
  __builtin_amdgcn_tensor_load_to_lds(g0, g1, z4, z4, z8, 0);
#else
  __builtin_amdgcn_tensor_load_to_lds(g0, g1, z4, z4, 0);
#endif
}
static __device__ __forceinline__ u32x4 tdm_g0(uint32_t lds_addr, uint64_t ga) {
  u32x4 g0;
  g0[0] = 1u;                                    // count=1, user descriptor
  g0[1] = lds_addr;                              // LDS byte address
  g0[2] = (uint32_t)ga;                          // global_addr[31:0]
  g0[3] = (uint32_t)((ga >> 32) & 0x01FFFFFFu)   // global_addr[56:32]
          | 0x80000000u;                         // type=2 ("image")
  return g0;
}
#endif

// ---------------------------------------------------------------------------
// Persistent LSTM kernel. 128 WGs: bb = blockIdx>>4 (16 batch rows),
// hs = blockIdx&15 (64 H cols per gate). 8 waves: wave w -> gate w>>1,
// column-pair w&1 (2 x 16-col WMMA tiles), K = 1152 split as 128 (x) + 1024 (h).
// Both LDS A-tiles staged by the Tensor Data Mover (waves 0 and 1 in parallel).
// ---------------------------------------------------------------------------
__global__ void __launch_bounds__(NTHR, 1)
lstm_persistent(const bf16*  __restrict__ xbf,
                const bf16*  __restrict__ Wf,
                const float* __restrict__ bias,
                bf16* __restrict__ hb0, bf16* __restrict__ hb1,
                float* __restrict__ cb,
                bf16* __restrict__ ob,
                const bf16*  __restrict__ Wlt,
                const float* __restrict__ bl,
                float* __restrict__ out_logits,
                float* __restrict__ out_h,
                float* __restrict__ out_c,
                int* __restrict__ cnt) {
  __shared__ bf16  Ax[16 * 128]  __attribute__((aligned(16)));   //  4 KB, packed
  __shared__ bf16  Ah[16 * 1024] __attribute__((aligned(16)));   // 32 KB, packed
  __shared__ float gbuf[4 * 16 * 64];                            // 16 KB

  const int tid  = threadIdx.x;
  const int bb   = blockIdx.x >> 4;
  const int hs   = blockIdx.x & 15;
  const int rowB = bb * 16;

  const int lane = tid & 31;
  const int wv   = tid >> 5;             // wave 0..7
  const int g    = wv >> 1;              // gate 0..3
  const int cp   = wv & 1;               // column pair 0..1
  const int ln   = lane & 15;            // B/C column, A row
  const int half = lane >> 4;

  // This wave's two weight columns (column-major [col][K]).
  const bf16* wcol0c = Wf + (size_t)(g * H_ + hs * 64 + cp * 32 + ln) * KDIM;
  const bf16* wcol1c = wcol0c + (size_t)16 * KDIM;

  for (int t = 0; t < T_; ++t) {
    const bf16* hread  = (t & 1) ? hb1 : hb0;
    bf16*       hwrite = (t & 1) ? hb0 : hb1;

    // Launder a zero element-offset (not the pointer itself): keeps the loads
    // rooted at the global kernel arg (=> global_load_b128, not flat_load) while
    // still preventing LICM from hoisting + spilling t-invariant fragments.
    int toff = 0;
    asm volatile("" : "+v"(toff));
    const bf16* wcol0 = wcol0c + toff;
    const bf16* wcol1 = wcol1c + toff;

#if defined(HAVE_TDM)
    if (wv == 0) {
      // h tile: flat 32 KB, 1-D copy -> Ah
      uint64_t ga = (uint64_t)(uintptr_t)(hread + (size_t)rowB * H_);
      i32x8 g1;
      g1[0] = 1 << 16;                 // data_size = 2 bytes
      g1[1] = (int)(16384u << 16);     // tensor_dim0 = 16384
      g1[2] = 1 << 16;                 // tensor_dim1 = 1
      g1[3] = (int)(16384u << 16);     // tile_dim0 = 16384
      g1[4] = 1;                       // tile_dim1 = 1
      g1[5] = 16384;                   // tensor_dim0_stride
      g1[6] = 0; g1[7] = 0;
      tdm_load(tdm_g0((uint32_t)(uintptr_t)Ah, ga), g1);
      __builtin_amdgcn_s_wait_tensorcnt(0);
    } else if (wv == 1) {
      // x tile: 16 rows x 128 elems, row stride T*V elements -> Ax (packed)
      uint64_t ga = (uint64_t)(uintptr_t)(xbf + ((size_t)rowB * T_ + t) * V_);
      i32x8 g1;
      g1[0] = 1 << 16;                 // data_size = 2 bytes
      g1[1] = (int)(128u << 16);       // tensor_dim0 = 128
      g1[2] = 16 << 16;                // tensor_dim1 = 16
      g1[3] = (int)(128u << 16);       // tile_dim0 = 128
      g1[4] = 16;                      // tile_dim1 = 16
      g1[5] = T_ * V_;                 // tensor_dim0_stride = 32768
      g1[6] = 0; g1[7] = 0;
      tdm_load(tdm_g0((uint32_t)(uintptr_t)Ax, ga), g1);
      __builtin_amdgcn_s_wait_tensorcnt(0);
    }
#else
    // Fallback: plain vector copies
    {
      int e = tid;                       // 256 x uint4 for Ax
      int r = e >> 4, qi = e & 15;
      ((uint4*)Ax)[e] =
          ((const uint4*)(xbf + ((size_t)(rowB + r) * T_ + t) * V_))[qi];
    }
#pragma unroll
    for (int i = 0; i < 8; ++i) {        // 2048 x uint4 for Ah
      int e = i * NTHR + tid;
      ((uint4*)Ah)[e] = ((const uint4*)(hread + (size_t)rowB * H_))[e];
    }
#endif
    __syncthreads();

    // ---- WMMA over K: phase 1 = x part (4 k-steps)
    v8f acc0 = v8f{}, acc1 = v8f{};
#pragma unroll
    for (int ks = 0; ks < 4; ++ks) {
      const int kb = ks * 32;
      ABf a;
      const bf16* ap = &Ax[ln * 128 + kb + half * 16];
      a.q[0] = *(const uint4*)ap;
      a.q[1] = *(const uint4*)(ap + 8);
      const int koff = kb + half * 16;
      ABf b0, b1;
      b0.q[0] = *(const uint4*)(wcol0 + koff);
      b0.q[1] = *(const uint4*)(wcol0 + koff + 8);
      b1.q[0] = *(const uint4*)(wcol1 + koff);
      b1.q[1] = *(const uint4*)(wcol1 + koff + 8);
      acc0 = wmma_bf16(a.v, b0.v, acc0);
      acc1 = wmma_bf16(a.v, b1.v, acc1);
      kstep_sched_hint();
    }

    // ---- phase 2 = h part (32 k-steps), 2-stage software pipeline
    {
      ABf a_c, b0_c, b1_c;
      {
        const bf16* ap = &Ah[ln * 1024 + half * 16];
        a_c.q[0] = *(const uint4*)ap;
        a_c.q[1] = *(const uint4*)(ap + 8);
        const int koff = 128 + half * 16;
        b0_c.q[0] = *(const uint4*)(wcol0 + koff);
        b0_c.q[1] = *(const uint4*)(wcol0 + koff + 8);
        b1_c.q[0] = *(const uint4*)(wcol1 + koff);
        b1_c.q[1] = *(const uint4*)(wcol1 + koff + 8);
      }
      for (int ks = 0; ks < 31; ++ks) {
        const int kb   = (ks + 1) * 32;
        const int koff = 128 + kb + half * 16;
        __builtin_prefetch((const void*)(wcol0 + koff + 32), 0, 1);
        ABf a_n, b0_n, b1_n;
        const bf16* ap = &Ah[ln * 1024 + kb + half * 16];
        a_n.q[0] = *(const uint4*)ap;
        a_n.q[1] = *(const uint4*)(ap + 8);
        b0_n.q[0] = *(const uint4*)(wcol0 + koff);
        b0_n.q[1] = *(const uint4*)(wcol0 + koff + 8);
        b1_n.q[0] = *(const uint4*)(wcol1 + koff);
        b1_n.q[1] = *(const uint4*)(wcol1 + koff + 8);
        acc0 = wmma_bf16(a_c.v, b0_c.v, acc0);
        acc1 = wmma_bf16(a_c.v, b1_c.v, acc1);
        kstep_sched_hint();
        a_c = a_n; b0_c = b0_n; b1_c = b1_n;
      }
      acc0 = wmma_bf16(a_c.v, b0_c.v, acc0);
      acc1 = wmma_bf16(a_c.v, b1_c.v, acc1);
    }

    // ---- exchange gate tiles through LDS: gbuf[gate][row][col64]
#pragma unroll
    for (int i = 0; i < 8; ++i) {
      gbuf[g * 1024 + (half * 8 + i) * 64 + cp * 32 + 0  + ln] = acc0[i];
      gbuf[g * 1024 + (half * 8 + i) * 64 + cp * 32 + 16 + ln] = acc1[i];
    }
    __syncthreads();

    // ---- activations + state update (4 cells per thread)
#pragma unroll
    for (int i = 0; i < 4; ++i) {
      int e  = i * NTHR + tid;       // 0..1023 = r*64 + j
      int r  = e >> 6, j = e & 63;
      int jg = hs * 64 + j;
      int b  = rowB + r;
      float gi = gbuf[0 * 1024 + e] + bias[0 * H_ + jg];
      float gf = gbuf[1 * 1024 + e] + bias[1 * H_ + jg];
      float gg = gbuf[2 * 1024 + e] + bias[2 * H_ + jg];
      float go = gbuf[3 * 1024 + e] + bias[3 * H_ + jg];
      float si = 1.f / (1.f + __expf(-gi));
      float sf = 1.f / (1.f + __expf(-gf));
      float tg = tanhf(gg);
      float so = 1.f / (1.f + __expf(-go));
      size_t ci = (size_t)b * H_ + jg;
      float cn = sf * cb[ci] + si * tg;
      float hn = so * tanhf(cn);
      cb[ci]     = cn;
      hwrite[ci] = (bf16)hn;
      ob[((size_t)b * T_ + t) * H_ + jg] = (bf16)so;   // reference collects o-gate
      if (t == T_ - 1) { out_h[ci] = hn; out_c[ci] = cn; }
    }

    grid_barrier(cnt, (t + 1) * NWG);
  }

  // ---- logits GEMM: [32768 x 1024](bf16 o) x [1024 x 128] -> f32 + bl
  // wave wv owns one 16-column tile; pipelined over 32 k-steps.
  for (int rt = blockIdx.x; rt < (B_ * T_) / 16; rt += NWG) {
    const int R = rt * 16;
    const bf16* arow = ob  + (size_t)(R + ln) * H_;
    const bf16* bcol = Wlt + (size_t)(wv * 16 + ln) * H_;
    v8f d = v8f{};
    ABf a_c, b_c;
    {
      const bf16* ap = arow + half * 16;
      a_c.q[0] = *(const uint4*)ap;
      a_c.q[1] = *(const uint4*)(ap + 8);
      const bf16* bp = bcol + half * 16;
      b_c.q[0] = *(const uint4*)bp;
      b_c.q[1] = *(const uint4*)(bp + 8);
    }
    for (int ks = 0; ks < 31; ++ks) {
      const int kb = (ks + 1) * 32;
      ABf a_n, b_n;
      const bf16* ap = arow + kb + half * 16;
      a_n.q[0] = *(const uint4*)ap;
      a_n.q[1] = *(const uint4*)(ap + 8);
      const bf16* bp = bcol + kb + half * 16;
      b_n.q[0] = *(const uint4*)bp;
      b_n.q[1] = *(const uint4*)(bp + 8);
      d = wmma_bf16(a_c.v, b_c.v, d);
      a_c = a_n; b_c = b_n;
    }
    d = wmma_bf16(a_c.v, b_c.v, d);
#pragma unroll
    for (int i = 0; i < 8; ++i) {
      int row = R + half * 8 + i;
      int c0  = wv * 16 + ln;
      out_logits[(size_t)row * V_ + c0] = d[i] + bl[c0];
    }
  }
}

// ---------------------------------------------------------------------------
// kernel_launch
// Inputs (dict order): x, h, c, Wx, Wh, bx, bh, Wl, bl
// Output: logits [B*T*V] f32 ++ h [B*H] f32 ++ c [B*H] f32
// ---------------------------------------------------------------------------
extern "C" void kernel_launch(void* const* d_in, const int* in_sizes, int n_in,
                              void* d_out, int out_size, void* d_ws, size_t ws_size,
                              hipStream_t stream) {
  const float* x  = (const float*)d_in[0];
  const float* h0 = (const float*)d_in[1];
  const float* c0 = (const float*)d_in[2];
  const float* Wx = (const float*)d_in[3];
  const float* Wh = (const float*)d_in[4];
  const float* bx = (const float*)d_in[5];
  const float* bh = (const float*)d_in[6];
  const float* Wl = (const float*)d_in[7];
  const float* bl = (const float*)d_in[8];

  float* out_logits = (float*)d_out;
  float* out_h      = out_logits + (size_t)B_ * T_ * V_;
  float* out_c      = out_h + (size_t)B_ * H_;

  // Workspace carve-out (256B aligned slabs), ~87 MB total.
  char*  base = (char*)d_ws;
  size_t off  = 0;
  auto carve = [&](size_t bytes) -> void* {
    void* p = base + off;
    off += (bytes + 255) & ~(size_t)255;
    return p;
  };
  bf16*  Wf   = (bf16*) carve((size_t)G4H * KDIM * sizeof(bf16));   // 9.0 MB
  bf16*  xbf  = (bf16*) carve((size_t)B_ * T_ * V_ * sizeof(bf16)); // 8.4 MB
  bf16*  Wlt  = (bf16*) carve((size_t)V_ * H_ * sizeof(bf16));      // 256 KB
  float* bias = (float*)carve((size_t)G4H * sizeof(float));         // 16 KB
  bf16*  hb0  = (bf16*) carve((size_t)B_ * H_ * sizeof(bf16));      // 256 KB
  bf16*  hb1  = (bf16*) carve((size_t)B_ * H_ * sizeof(bf16));      // 256 KB
  float* cb   = (float*)carve((size_t)B_ * H_ * sizeof(float));     // 512 KB
  bf16*  ob   = (bf16*) carve((size_t)B_ * T_ * H_ * sizeof(bf16)); // 64 MB
  int*   cnt  = (int*)  carve(256);

  {
    size_t n = (size_t)G4H * KDIM;
    prep_weights<<<(unsigned)((n + 255) / 256), 256, 0, stream>>>(Wx, Wh, Wf);
  }
  {
    size_t n = (size_t)B_ * T_ * V_;
    prep_x<<<(unsigned)((n + 255) / 256), 256, 0, stream>>>(x, xbf);
  }
  prep_misc<<<(B_ * H_ + 255) / 256, 256, 0, stream>>>(bx, bh, Wl, h0, c0,
                                                       bias, Wlt, hb0, cb, cnt);
  lstm_persistent<<<NWG, NTHR, 0, stream>>>(xbf, Wf, bias, hb0, hb1, cb, ob,
                                            Wlt, bl, out_logits, out_h, out_c, cnt);
}
